// _TorchWLKernel_63900523430226
// MI455X (gfx1250) — matrix operand order repaired and checked by reading, not compiled, not run on hardware
//
#include <hip/hip_runtime.h>
#include <hip/hip_bf16.h>

// WL graph kernel for MI455X (gfx1250, wave32).
// Pipeline per call (all on `stream`, graph-capture safe):
//   zero ws -> copy labels -> 4x { histogram(atomic f32) -> Gram via TDM-fed
//   WMMA f16 (self-zeroing counts) } interleaved with 3x { edge hash scatter
//   (atomic u64) -> relabel(splitmix64 into [0,total)) } -> normalize -> d_out.
// Workspace: Kacc(4KB) | cur labels(1MB) | nb_hash u64(2MB) | counts f32 (32MB)
//   ~36.7MB total, entirely L2-resident on MI455X (192MB L2).

typedef __attribute__((ext_vector_type(16))) _Float16     v16h;
typedef __attribute__((ext_vector_type(8)))  float        v8f;
typedef __attribute__((ext_vector_type(4)))  unsigned int v4u;
typedef __attribute__((ext_vector_type(8)))  int          v8i;
typedef __attribute__((ext_vector_type(4)))  int          v4i;

__device__ __forceinline__ unsigned long long wl_mix64(unsigned long long x) {
  x += 0x9E3779B97F4A7C15ull;
  x = (x ^ (x >> 30)) * 0xBF58476D1CE4E5B9ull;
  x = (x ^ (x >> 27)) * 0x94D049BB133111EBull;
  return x ^ (x >> 31);
}

__global__ void wl_zero_u32(unsigned int* p, long long n) {
  long long i = (long long)blockIdx.x * blockDim.x + threadIdx.x;
  long long s = (long long)gridDim.x * blockDim.x;
  for (; i < n; i += s) p[i] = 0u;
}

__global__ void wl_copy_labels(const int* __restrict__ labels, int* __restrict__ cur,
                               long long n) {
  long long i = (long long)blockIdx.x * blockDim.x + threadIdx.x;
  long long s = (long long)gridDim.x * blockDim.x;
  for (; i < n; i += s) cur[i] = labels[i];
}

// Histogram: counts[label][graph] += weight. Scattered f32 atomics, L2-resident.
__global__ void wl_hist(const int* __restrict__ cur, const float* __restrict__ w,
                        float* __restrict__ counts, int B, int N, long long total) {
  long long i = (long long)blockIdx.x * blockDim.x + threadIdx.x;
  long long s = (long long)gridDim.x * blockDim.x;
  for (; i < total; i += s) {
    if (i + s < total) {  // stream one grid-stride ahead (global_prefetch_b8)
      __builtin_prefetch(&cur[i + s], 0, 1);
      __builtin_prefetch(&w[i + s], 0, 1);
    }
    int b = (int)(i / N);
    int lbl = cur[i];
    atomicAdd(&counts[(long long)lbl * B + b], w[i]);
  }
}

// ---------------------------------------------------------------------------
// Tensor Data Mover: DMA one 32x32 f32 tile of counts into LDS.
// D# per CDNA5 ISA: group0 = {count=1, lds_addr,
// global_addr[56:0], type=2}; group1 = {data_size=4B, tensor_dim0=32,
// tensor_dim1=rows (tail rows auto-zero via OOB), tile_dim0=32, tile_dim1=32,
// tensor_dim0_stride=32}. Groups 2/3 zero (2-D tensor). Tracked by TENSORcnt.
// ---------------------------------------------------------------------------
__device__ __forceinline__ void wl_tdm_load_tile(unsigned lds_addr,
                                                 unsigned long long gaddr,
                                                 unsigned tensor_rows) {
  v4u g0;
  g0[0] = 1u;                                                   // count=1, user D#
  g0[1] = lds_addr;                                             // bits 63:32
  g0[2] = (unsigned)gaddr;                                      // bits 95:64
  g0[3] = (unsigned)((gaddr >> 32) & 0x01FFFFFFull) | (2u << 30); // type=2
  v8i g1;
  g1[0] = (int)(2u << 16);            // workgroup_mask=0, data_size=2 (4 bytes)
  g1[1] = (int)(32u << 16);           // tensor_dim0 = 32 (graphs per row)
  g1[2] = (int)(tensor_rows << 16);   // tensor_dim1[15:0] (rows in bounds)
  g1[3] = (int)(32u << 16);           // tile_dim0 = 32
  g1[4] = 32;                         // tile_dim1 = 32 (OOB rows read as zero)
  g1[5] = 32;                         // tensor_dim0_stride = 32 elements
  g1[6] = 0;
  g1[7] = 0;
  v4i zz = {0, 0, 0, 0};
#if defined(__clang_major__) && (__clang_major__ >= 23)
  v8i z8 = {0, 0, 0, 0, 0, 0, 0, 0};
  __builtin_amdgcn_tensor_load_to_lds(g0, g1, zz, zz, z8, 0);
#else
  __builtin_amdgcn_tensor_load_to_lds(g0, g1, zz, zz, 0);
#endif
}

// K = H * H^T for B == 32 via v_wmma_f32_16x16x32_f16, fed by the TDM.
// Each wave owns disjoint 32-label chunks. Double-buffered per-wave LDS tiles:
// issue TDM for chunk t+1, s_wait_tensorcnt(1) for chunk t (TDM in-order per
// wave), zero the exclusively-owned global chunk for the next WL iteration,
// swizzle A/B per ISA 7.12.2 with f32->f16 convert on LDS read, 4 WMMAs for
// the 2x2 tiling of the 32x32 Gram output, atomically reduce into Kacc.
#define WL_WAVES 8
__global__ void __launch_bounds__(256) wl_gram_wmma(float* __restrict__ counts,
                                                    float* __restrict__ Kacc,
                                                    long long total) {
  __shared__ float tile[2][WL_WAVES][32][32];   // 64KB of the 320KB WGP LDS
  const int lane  = threadIdx.x & 31;
  const int waveu = __builtin_amdgcn_readfirstlane((int)(threadIdx.x >> 5));
  const int m  = lane & 15;   // row-in-tile (A) / col-in-tile (B, C/D)
  const int hi = lane >> 4;   // lane half selects K sub-range / M half of C/D

  const long long nchunks = (total + 31) >> 5;
  const long long slot    = (long long)blockIdx.x * WL_WAVES + waveu;
  const long long nslots  = (long long)gridDim.x * WL_WAVES;
  const long long iters   = (nchunks + nslots - 1) / nslots;
  const long long safec   = (slot < nchunks) ? slot : 0;  // dummy-load target

  unsigned ldsA[2];
  ldsA[0] = (unsigned)(unsigned long long)&tile[0][waveu][0][0];
  ldsA[1] = (unsigned)(unsigned long long)&tile[1][waveu][0][0];

  // Prologue: start DMA of the wave's first chunk into buffer 0.
  {
    const long long l0 = safec << 5;
    const long long rem = total - l0;
    wl_tdm_load_tile(ldsA[0],
                     (unsigned long long)(unsigned long long)(counts + l0 * 32),
                     (unsigned)(rem < 32 ? rem : 32));
  }

  v8f acc00 = {}, acc01 = {}, acc10 = {}, acc11 = {};

  for (long long t = 0; t < iters; ++t) {
    const int cur = (int)(t & 1);
    const long long chunk = slot + t * nslots;

    // Kick the next chunk's DMA into the other buffer (dummy when past end so
    // the wait immediate below stays a compile-time constant).
    {
      const long long nc = chunk + nslots;
      const long long c  = (nc < nchunks) ? nc : safec;
      const long long l0 = c << 5;
      const long long rem = total - l0;
      wl_tdm_load_tile(ldsA[cur ^ 1],
                       (unsigned long long)(unsigned long long)(counts + l0 * 32),
                       (unsigned)(rem < 32 ? rem : 32));
    }
    __builtin_amdgcn_s_wait_tensorcnt(1);  // current tile resident in LDS
    asm volatile("" ::: "memory");

    if (chunk < nchunks) {                 // wave-uniform: EXEC full for WMMA
      const long long l0  = chunk << 5;
      const long long row = l0 + lane;
      if (row < total) {
        // chunk is exclusively ours and already in LDS: clear it for the next
        // WL iteration's histogram (overlaps with the in-flight TDM).
        float* dst = counts + row * 32;
#pragma unroll
        for (int c = 0; c < 32; ++c) dst[c] = 0.0f;
      }

      // A 16x32 f16 (ISA 7.12.2): lane<16 holds K 0-7 (e0-7), 16-23 (e8-15);
      // lane>=16 holds K 8-15, 24-31. B 32x16: same K order as A;
      // n = lane&15. f32->f16 convert on LDS read.
      const volatile float* tp = &tile[cur][waveu][0][0];
      v16h a0, a1, b0, b1;
#pragma unroll
      for (int e = 0; e < 16; ++e) {
        const int ka = (e & 7) + ((e >> 3) << 4) + (hi << 3);
        const int kb = (e & 7) + ((e >> 3) << 4) + (hi << 3);
        a0[e] = (_Float16)tp[ka * 32 + m];         // graphs i = 0..15
        a1[e] = (_Float16)tp[ka * 32 + 16 + m];    // graphs i = 16..31
        b0[e] = (_Float16)tp[kb * 32 + m];         // graphs j = 0..15
        b1[e] = (_Float16)tp[kb * 32 + 16 + m];    // graphs j = 16..31
      }
      acc00 = __builtin_amdgcn_wmma_f32_16x16x32_f16(false, a0, false, b0,
                                                     (short)0, acc00, false, false);
      acc01 = __builtin_amdgcn_wmma_f32_16x16x32_f16(false, a0, false, b1,
                                                     (short)0, acc01, false, false);
      acc10 = __builtin_amdgcn_wmma_f32_16x16x32_f16(false, a1, false, b0,
                                                     (short)0, acc10, false, false);
      acc11 = __builtin_amdgcn_wmma_f32_16x16x32_f16(false, a1, false, b1,
                                                     (short)0, acc11, false, false);
    }
  }

  // C/D layout: VGPR r -> M = r + 8*hi, N = lane&15. Reduce across waves.
#pragma unroll
  for (int r = 0; r < 8; ++r) {
    const int i = (hi << 3) + r;
    const int j = m;
    atomicAdd(&Kacc[i * 32 + j],               acc00[r]);
    atomicAdd(&Kacc[i * 32 + (16 + j)],        acc01[r]);
    atomicAdd(&Kacc[(16 + i) * 32 + j],        acc10[r]);
    atomicAdd(&Kacc[(16 + i) * 32 + (16 + j)], acc11[r]);
  }
}

// Generic fallback Gram (never used for the reference shape B==32).
__global__ void wl_gram_fallback(const float* __restrict__ counts,
                                 float* __restrict__ Kacc, int B, long long total) {
  const int i = blockIdx.x / B, j = blockIdx.x % B;
  float s = 0.0f;
  for (long long l = threadIdx.x; l < total; l += blockDim.x)
    s += counts[l * B + i] * counts[l * B + j];
  __shared__ float red[256];
  red[threadIdx.x] = s;
  __syncthreads();
  for (int off = 128; off > 0; off >>= 1) {
    if ((int)threadIdx.x < off) red[threadIdx.x] += red[threadIdx.x + off];
    __syncthreads();
  }
  if (threadIdx.x == 0) atomicAdd(&Kacc[i * B + j], red[0]);
}

// Per-edge neighbor-hash scatter: nb_hash[row] += mix64(seed ^ label[col]).
// 4.2M global_atomic_add_u64 per iteration, all hitting a 2MB L2-resident array.
__global__ void wl_scatter(const int* __restrict__ edges, const int* __restrict__ cur,
                           unsigned long long* __restrict__ nbh,
                           int B, int E, int N, unsigned long long seed) {
  long long i = (long long)blockIdx.x * blockDim.x + threadIdx.x;
  const long long s = (long long)gridDim.x * blockDim.x;
  const long long tot = (long long)B * E;
  for (; i < tot; i += s) {
    const int b = (int)(i / E);
    const int e = (int)(i - (long long)b * E);
    const int* eb = edges + (long long)b * 2 * E;
    const int row = eb[e];
    const int col = eb[E + e];
    const unsigned long long h =
        wl_mix64(seed ^ (unsigned long long)(unsigned int)cur[(long long)b * N + col]);
    atomicAdd(&nbh[(long long)b * N + row], h);
  }
}

// Relabel: raw = own_hash + neighbor_multiset_hash; map into [0,total) by a
// second mix (stands in for jnp.unique compaction; same id-space width).
// Also resets nb_hash for the next scatter pass.
__global__ void wl_relabel(int* __restrict__ cur, unsigned long long* __restrict__ nbh,
                           long long total, unsigned long long seed_own) {
  long long i = (long long)blockIdx.x * blockDim.x + threadIdx.x;
  const long long s = (long long)gridDim.x * blockDim.x;
  const bool pow2 = (total & (total - 1)) == 0;
  for (; i < total; i += s) {
    const unsigned long long own =
        wl_mix64(seed_own ^ (unsigned long long)(unsigned int)cur[i]);
    const unsigned long long raw = own + nbh[i];
    nbh[i] = 0ull;
    const unsigned long long m2 = wl_mix64(raw);
    cur[i] = (int)(pow2 ? (m2 & (unsigned long long)(total - 1))
                        : (m2 % (unsigned long long)total));
  }
}

__global__ void wl_normalize(const float* __restrict__ Kacc, float* __restrict__ out,
                             int B) {
  int t = blockIdx.x * blockDim.x + threadIdx.x;
  const int n = B * B;
  for (; t < n; t += gridDim.x * blockDim.x) {
    const int i = t / B, j = t % B;
    const float dii = Kacc[i * B + i];
    const float djj = Kacc[j * B + j];
    out[t] = Kacc[t] * rsqrtf(dii * djj);
  }
}

extern "C" void kernel_launch(void* const* d_in, const int* in_sizes, int n_in,
                              void* d_out, int out_size, void* d_ws, size_t ws_size,
                              hipStream_t stream) {
  const int*   edges   = (const int*)d_in[0];
  const int*   labels  = (const int*)d_in[1];
  const float* weights = (const float*)d_in[2];
  float* out = (float*)d_out;

  int B = 1;
  while ((long long)B * B < (long long)out_size) ++B;   // out is B x B
  const long long total = in_sizes[1];                  // B * N
  const int N = (int)(total / B);
  const int E = in_sizes[0] / (2 * B);

  char* ws = (char*)d_ws;
  const size_t oK   = 0;
  const size_t oCur = (oK + (size_t)B * B * sizeof(float) + 255) & ~(size_t)255;
  const size_t oNbh = (oCur + (size_t)total * sizeof(int) + 255) & ~(size_t)255;
  const size_t oCnt = (oNbh + (size_t)total * sizeof(unsigned long long) + 255) & ~(size_t)255;
  const size_t oEnd = oCnt + (size_t)total * (size_t)B * sizeof(float);

  float*              Kacc   = (float*)(ws + oK);
  int*                cur    = (int*)(ws + oCur);
  unsigned long long* nbh    = (unsigned long long*)(ws + oNbh);
  float*              counts = (float*)(ws + oCnt);

  // Deterministic per call: clear all workspace state up front.
  wl_zero_u32<<<4096, 256, 0, stream>>>((unsigned int*)ws, (long long)(oEnd / 4));
  wl_copy_labels<<<1024, 256, 0, stream>>>(labels, cur, total);

  const int N_ITER = 3;
  for (int it = 0; it <= N_ITER; ++it) {
    wl_hist<<<2048, 256, 0, stream>>>(cur, weights, counts, B, N, total);
    if (B == 32 && total > 0) {
      wl_gram_wmma<<<128, 256, 0, stream>>>(counts, Kacc, total);  // self-zeros counts
    } else {
      wl_gram_fallback<<<B * B, 256, 0, stream>>>(counts, Kacc, B, total);
      wl_zero_u32<<<4096, 256, 0, stream>>>((unsigned int*)counts,
                                            (long long)total * B);
    }
    if (it < N_ITER) {
      const unsigned long long s_nb =
          0xA24BAED4963EE407ull + 0x9E3779B97F4A7C15ull * (unsigned long long)it;
      const unsigned long long s_own =
          0x452821E638D01377ull + 0xD1B54A32D192ED03ull * (unsigned long long)it;
      wl_scatter<<<8192, 256, 0, stream>>>(edges, cur, nbh, B, E, N, s_nb);
      wl_relabel<<<1024, 256, 0, stream>>>(cur, nbh, total, s_own);
    }
  }
  wl_normalize<<<(out_size + 255) / 256, 256, 0, stream>>>(Kacc, out, B);
  (void)n_in; (void)ws_size; (void)weights;
}
